// RadixAttention_67233418052078
// MI455X (gfx1250) — matrix operand order repaired and checked by reading, not compiled, not run on hardware
//
#include <hip/hip_runtime.h>

// ---- CDNA5 WMMA vector types -------------------------------------------------
typedef __attribute__((ext_vector_type(16))) __bf16 v16bf;
typedef __attribute__((ext_vector_type(8)))  float  v8f;

namespace {
constexpr int B_  = 2;
constexpr int E_  = 1024;
constexpr int P_  = 1024;
constexpr int HQ_ = 32;
constexpr int HKV_= 8;
constexpr int D_  = 128;
constexpr int G_  = 4;                       // HQ / HKV
constexpr int TK  = 32;                      // keys per KV tile (WMMA K for PV)
constexpr int ROWSTRIDE = HKV_ * D_;         // floats between tokens in k/v layouts
constexpr float SCALE_ = 0.08838834764831845f;    // 1/sqrt(128)
constexpr float CAP_   = 30.0f;
constexpr float QPRE_  = SCALE_ / CAP_;           // folded into Q at load
constexpr float CAPL2_ = CAP_ * 1.4426950408889634f; // cap * log2(e): base-2 softmax
}

// pack two f32 -> bf16x2 via hardware conversion (v_cvt_pk_bf16_f32 path)
__device__ __forceinline__ unsigned bf16pk(float a, float b) {
  union { __bf16 h[2]; unsigned u; } t;
  t.h[0] = (__bf16)a;
  t.h[1] = (__bf16)b;
  return t.u;
}
__device__ __forceinline__ unsigned short bf16b(float a) {
  union { __bf16 h; unsigned short u; } t;
  t.h = (__bf16)a;
  return t.u;
}

// native single-instruction transcendentals
__device__ __forceinline__ float fast_tanh(float x) {
#if __has_builtin(__builtin_amdgcn_tanhf)
  return __builtin_amdgcn_tanhf(x);
#else
  float y;
  // v_tanh_f32 is a TRANS op; v_nop covers the co-execution RAW hazard (ISA 7.4)
  asm volatile("v_tanh_f32 %0, %1\n\tv_nop" : "=v"(y) : "v"(x));
  return y;
#endif
}
__device__ __forceinline__ float fast_exp2(float x) {
#if __has_builtin(__builtin_amdgcn_exp2f)
  return __builtin_amdgcn_exp2f(x);
#else
  return exp2f(x);
#endif
}

union V16 { uint4 q[2]; unsigned u[8]; v16bf v; };

__global__ __launch_bounds__(128, 1)
void radix_attn_fused(const float* __restrict__ q,
                      const float* __restrict__ k,
                      const float* __restrict__ v,
                      const float* __restrict__ pkv_k,
                      const float* __restrict__ pkv_v,
                      float* __restrict__ out) {
  // LDS: K tile bf16 [buf][32 keys][128 d], V tile bf16 transposed [buf][128 d][32 keys]
  __shared__ __align__(16) unsigned sK [2][TK * (D_ / 2)];   // 2 x 8KB
  __shared__ __align__(16) unsigned sVT[2][D_ * (TK / 2)];   // 2 x 8KB

  const int tid  = threadIdx.x;
  const int lane = tid & 31;
  const int wave = tid >> 5;
  const int n    = lane & 15;   // WMMA column (= q-row id within wave tile)
  const int h    = lane >> 4;   // half-wave

  const int bid   = blockIdx.x;
  const int etile = bid & 63;          // 64 tiles of 16 e each
  const int hkv   = (bid >> 6) & 7;
  const int b     = bid >> 9;

  const int e0w    = etile * 16 + wave * 4;   // this wave: e0w .. e0w+3 (x4 g)
  const int e_lane = e0w + (n >> 2);
  const int g      = n & 3;

  // ---- Q tile resident in registers, B-operand layout (N=n, K=d), pre-scaled ----
  V16 qB[4];
  {
    const float* qrow =
        q + ((size_t)((b * E_ + e_lane) * HQ_) + hkv * G_ + g) * D_;
    #pragma unroll
    for (int c = 0; c < 4; ++c) {
      const float* p0 = qrow + c * 32 + h * 8;    // element kk -> d = 32c+8h+kk (kk<8), 32c+16+8h+(kk-8) (kk>=8)
      float f[16];
      #pragma unroll
      for (int i = 0; i < 4; ++i) {
        float4 t = ((const float4*)p0)[i + (i >> 1) * 2];
        f[4*i+0] = t.x * QPRE_; f[4*i+1] = t.y * QPRE_;
        f[4*i+2] = t.z * QPRE_; f[4*i+3] = t.w * QPRE_;
      }
      #pragma unroll
      for (int j = 0; j < 8; ++j) qB[c].u[j] = bf16pk(f[2*j], f[2*j+1]);
    }
  }

  // ---- online softmax state (base-2 domain) ----
  float m_run = -INFINITY, l_run = 0.0f;
  v8f acc[8];
  #pragma unroll
  for (int dt = 0; dt < 8; ++dt)
    #pragma unroll
    for (int i = 0; i < 8; ++i) acc[dt][i] = 0.0f;

  const float* kbase_pref = pkv_k + (size_t)b * P_ * ROWSTRIDE + hkv * D_;
  const float* vbase_pref = pkv_v + (size_t)b * P_ * ROWSTRIDE + hkv * D_;
  const float* kbase_new  = k     + (size_t)b * E_ * ROWSTRIDE + hkv * D_;
  const float* vbase_new  = v     + (size_t)b * E_ * ROWSTRIDE + hkv * D_;

  const int pref_tiles = P_ / TK;                      // 32
  const int e_blk_max  = etile * 16 + 15;
  const int ntiles     = pref_tiles + (e_blk_max >> 5) + 1;  // skip fully-masked

  const int ldr  = tid >> 2;          // key row loaded by this thread (0..31)
  const int dseg = (tid & 3) * 32;    // 32-wide d segment

  for (int kt = 0; kt < ntiles; ++kt) {
    const int  buf    = kt & 1;
    const bool is_new = (kt >= pref_tiles);
    const int  t0     = (is_new ? (kt - pref_tiles) : kt) * TK;
    const float* kb   = is_new ? kbase_new : kbase_pref;
    const float* vb   = is_new ? vbase_new : vbase_pref;

    // ---- cooperative stage: K -> sK (row major bf16), V -> sVT (transposed) ----
    {
      const float* kr = kb + (size_t)(t0 + ldr) * ROWSTRIDE + dseg;
      unsigned* dstk = &sK[buf][ldr * (D_ / 2) + (dseg >> 1)];
      #pragma unroll
      for (int i = 0; i < 4; ++i) {
        float4 a  = ((const float4*)kr)[2 * i];
        float4 bb = ((const float4*)kr)[2 * i + 1];
        uint4 pk4;
        pk4.x = bf16pk(a.x, a.y);  pk4.y = bf16pk(a.z, a.w);
        pk4.z = bf16pk(bb.x, bb.y); pk4.w = bf16pk(bb.z, bb.w);
        ((uint4*)dstk)[i] = pk4;
      }
      const float* vr = vb + (size_t)(t0 + ldr) * ROWSTRIDE + dseg;
      unsigned short* vt = (unsigned short*)&sVT[buf][0];
      #pragma unroll
      for (int i = 0; i < 8; ++i) {
        float4 a = ((const float4*)vr)[i];
        const int d = dseg + 4 * i;
        vt[(d + 0) * TK + ldr] = bf16b(a.x);
        vt[(d + 1) * TK + ldr] = bf16b(a.y);
        vt[(d + 2) * TK + ldr] = bf16b(a.z);
        vt[(d + 3) * TK + ldr] = bf16b(a.w);
      }
    }
    __syncthreads();   // one barrier/iter is safe with double buffering

    // ---- S^T = K_tile * Q^T : two 16-key subtiles, chained over d (K=32 x4) ----
    v8f s0, s1;
    #pragma unroll
    for (int i = 0; i < 8; ++i) { s0[i] = 0.0f; s1[i] = 0.0f; }
    const unsigned* kbuf = &sK[buf][0];
    #pragma unroll
    for (int c = 0; c < 4; ++c) {
      V16 a0, a1;
      const int base = c * 16 + h * 4;
      a0.q[0] = *(const uint4*)&kbuf[(n      ) * (D_/2) + base    ];
      a0.q[1] = *(const uint4*)&kbuf[(n      ) * (D_/2) + base + 8];
      a1.q[0] = *(const uint4*)&kbuf[(n + 16 ) * (D_/2) + base    ];
      a1.q[1] = *(const uint4*)&kbuf[(n + 16 ) * (D_/2) + base + 8];
      s0 = __builtin_amdgcn_wmma_f32_16x16x32_bf16(false, a0.v, false, qB[c].v,
                                                   (short)0, s0, false, false);
      s1 = __builtin_amdgcn_wmma_f32_16x16x32_bf16(false, a1.v, false, qB[c].v,
                                                   (short)0, s1, false, false);
    }

    // ---- soft-cap (native v_tanh) + causal mask, base-2 score domain ----
    float sc0[8], sc1[8];
    #pragma unroll
    for (int r = 0; r < 8; ++r) {
      float x0 = CAPL2_ * fast_tanh(s0[r]);   // Q pre-scaled by SCALE/CAP
      float x1 = CAPL2_ * fast_tanh(s1[r]);
      if (is_new) {
        const int tk0 = t0 + r + 8 * h;
        x0 = (tk0      > e_lane) ? -INFINITY : x0;
        x1 = (tk0 + 16 > e_lane) ? -INFINITY : x1;
      }
      sc0[r] = x0; sc1[r] = x1;
    }

    // ---- online softmax update (per q-row; lane + partner half-wave) ----
    float tm = -INFINITY;
    #pragma unroll
    for (int r = 0; r < 8; ++r) tm = fmaxf(tm, fmaxf(sc0[r], sc1[r]));
    tm = fmaxf(tm, __shfl_xor(tm, 16, 32));
    const float m_new = fmaxf(m_run, tm);
    const float corr  = fast_exp2(m_run - m_new);
    float p0[8], p1[8], lsum = 0.0f;
    #pragma unroll
    for (int r = 0; r < 8; ++r) {
      p0[r] = fast_exp2(sc0[r] - m_new);
      p1[r] = fast_exp2(sc1[r] - m_new);
      lsum += p0[r] + p1[r];
    }
    lsum += __shfl_xor(lsum, 16, 32);
    l_run = l_run * corr + lsum;
    m_run = m_new;
    #pragma unroll
    for (int dt = 0; dt < 8; ++dt)
      #pragma unroll
      for (int i = 0; i < 8; ++i) acc[dt][i] *= corr;

    // ---- rearrange P^T (C layout) into WMMA B-operand layout (keys x qrows) ----
    float lowv[8], highv[8];
    #pragma unroll
    for (int r = 0; r < 8; ++r) {
      const float o0 = __shfl_xor(p0[r], 16, 32);
      const float o1 = __shfl_xor(p1[r], 16, 32);
      lowv[r]  = h ? o1    : p0[r];   // key = 16h + r
      highv[r] = h ? p1[r] : o0;      // key = 16h + 8 + r
    }
    V16 pB;
    #pragma unroll
    for (int j = 0; j < 4; ++j) {
      pB.u[j]     = bf16pk(lowv [2*j], lowv [2*j+1]);
      pB.u[4 + j] = bf16pk(highv[2*j], highv[2*j+1]);
    }

    // ---- O^T += V^T * P^T : 8 d-tiles, K=32 keys each ----
    const unsigned* vtb = &sVT[buf][0];
    #pragma unroll
    for (int dt = 0; dt < 8; ++dt) {
      V16 av;
      const int d = dt * 16 + n;
      av.q[0] = *(const uint4*)&vtb[d * (TK/2) + 8 * h    ];
      av.q[1] = *(const uint4*)&vtb[d * (TK/2) + 8 * h + 4];
      acc[dt] = __builtin_amdgcn_wmma_f32_16x16x32_bf16(false, av.v, false, pB.v,
                                                        (short)0, acc[dt], false, false);
    }
  }

  // ---- epilogue: O = O^T / l, two b128 stores per d-tile ----
  const float rl = 1.0f / l_run;
  float* orow = out + ((size_t)((b * E_ + e_lane) * HQ_) + hkv * G_ + g) * D_;
  #pragma unroll
  for (int dt = 0; dt < 8; ++dt) {
    const int d0 = dt * 16 + 8 * h;   // 8 consecutive d per lane
    float4 lo = make_float4(acc[dt][0]*rl, acc[dt][1]*rl, acc[dt][2]*rl, acc[dt][3]*rl);
    float4 hi = make_float4(acc[dt][4]*rl, acc[dt][5]*rl, acc[dt][6]*rl, acc[dt][7]*rl);
    *(float4*)(orow + d0)     = lo;
    *(float4*)(orow + d0 + 4) = hi;
  }
}

extern "C" void kernel_launch(void* const* d_in, const int* in_sizes, int n_in,
                              void* d_out, int out_size, void* d_ws, size_t ws_size,
                              hipStream_t stream) {
  (void)in_sizes; (void)n_in; (void)out_size; (void)d_ws; (void)ws_size;
  const float* q  = (const float*)d_in[0];
  const float* k  = (const float*)d_in[1];
  const float* v  = (const float*)d_in[2];
  const float* pk = (const float*)d_in[3];
  const float* pv = (const float*)d_in[4];
  float* out = (float*)d_out;

  const int grid = B_ * HKV_ * (E_ / 16);   // 1024 blocks of 4 waves
  radix_attn_fused<<<dim3(grid), dim3(128), 0, stream>>>(q, k, v, pk, pv, out);
}
